// BilinearLabelAttention_56049323213771
// MI455X (gfx1250) — matrix operand order repaired and checked by reading, not compiled, not run on hardware
//
#include <hip/hip_runtime.h>

// out[b,l,i,o] = sum_j head[b,i,j] * U[l,j] * dep[b,o,j]
// Per (b,l): C = (head[b] * u_l) @ dep[b]^T  -- 512x512x512 GEMM on
// v_wmma_f32_16x16x32_bf16 with hi/lo bf16 split (3 WMMAs/tile, ~2^-17 rel).
//
// Round-3:
//   kernel 1: split dep -> bf16 hi/lo in d_ws, in WMMA-native 2KB chunks
//             [b][oBlk16][kBlk32]{hi 16x32, lo 16x32} so a wave's B-fragment
//             read is one contiguous 1KB stream (full 128B-line utilization).
//   kernel 2: block = (b, l, i-tile 128). Converts (head*u_l) once into 260KB
//             LDS, then sweeps O=512; A frags from LDS, B frags from tiled ws.
//   fallback: round-1 fused kernel if ws_size too small.

typedef __attribute__((ext_vector_type(16))) __bf16 v16bf;
typedef __attribute__((ext_vector_type(8)))  __bf16 bf16x8;
typedef __attribute__((ext_vector_type(8)))  float  v8f;
typedef __attribute__((ext_vector_type(4)))  float  f32x4;

#define CAT16(a, b) __builtin_shufflevector(a, b, 0,1,2,3,4,5,6,7,8,9,10,11,12,13,14,15)
#define WMMA_BF16(A, B, C) __builtin_amdgcn_wmma_f32_16x16x32_bf16( \
    false, (A), false, (B), (short)0, (C), false, false)

constexpr int B_ = 8, S_ = 512, D_ = 512, L_ = 32;
constexpr int TI = 128;          // i-rows per block
constexpr int TK = 32;           // K per WMMA
constexpr int LDA = 520;         // LDS row stride (bf16): 1040B, 16B-aligned,
                                 // rows advance 4 banks -> 16 rows distinct groups
constexpr int THREADS = 256;     // 8 wave32s
constexpr int CHUNK = 1024;      // bf16 elems per (oBlk,kBlk) chunk: 512 hi + 512 lo
constexpr size_t DEP_ELEMS = (size_t)B_ * S_ * D_;
constexpr size_t SMEM_BYTES = (size_t)TI * LDA * 2 * 2;   // 266,240B

__device__ __forceinline__ void split_bf16(float x, __bf16& h, __bf16& l) {
    h = (__bf16)x;
    l = (__bf16)(x - (float)h);
}

// ---- pre-pass: dep -> tiled hi/lo bf16 ---------------------------------------
// chunk c = (b*32 + oBlk)*16 + kBlk ; element  c*1024 + orow*32 + kk (+512 for lo)
__global__ __launch_bounds__(256)
void dep_split_tiled_kernel(const float* __restrict__ dep, __bf16* __restrict__ dT)
{
    const int tid  = threadIdx.x;
    const int b    = blockIdx.x >> 5;      // 0..7
    const int oBlk = blockIdx.x & 31;      // 0..31  (16 dep rows each)
    const float* __restrict__ src = dep + ((size_t)(b * S_) + oBlk * 16) * D_;
    const size_t cRow = ((size_t)(b * 32 + oBlk)) * 16;   // chunk base index

    #pragma unroll
    for (int q = 0; q < 8; ++q) {
        const int f    = q * 256 + tid;        // float4 id within 16x512 tile
        const int orow = f >> 7;               // 128 float4 per row
        const int kc   = (f & 127) << 2;
        const int kBlk = kc >> 5;
        const int kk   = kc & 31;
        f32x4 v = *(const f32x4*)(src + (size_t)orow * D_ + kc);
        __bf16* p = dT + (cRow + kBlk) * CHUNK + orow * 32 + kk;
        #pragma unroll
        for (int e = 0; e < 4; ++e) {
            __bf16 h, l2;
            split_bf16(v[e], h, l2);
            p[e]       = h;
            p[e + 512] = l2;
        }
    }
}

// ---- main kernel -------------------------------------------------------------
__global__ __launch_bounds__(THREADS)
void bla_main_kernel(const float* __restrict__ head,
                     const float* __restrict__ U,
                     const __bf16* __restrict__ depT,
                     float* __restrict__ out)
{
    extern __shared__ __align__(16) char smem[];
    __bf16 (* __restrict__ sAhi)[LDA] = (__bf16(*)[LDA])smem;
    __bf16 (* __restrict__ sAlo)[LDA] = (__bf16(*)[LDA])(smem + (size_t)TI * LDA * 2);

    const int tid   = threadIdx.x;
    const int iBase = blockIdx.x * TI;     // 0,128,256,384
    const int l     = blockIdx.y;
    const int b     = blockIdx.z;

    // Phase 1: A = head[b, iBase:+128, :] * u_l -> hi/lo bf16 in LDS (once).
    const int kc = (tid & 127) << 2;       // fixed 4 k-columns per thread
    const f32x4 u4 = *(const f32x4*)(U + l * D_ + kc);
    const float* __restrict__ headB = head + ((size_t)b * S_ + iBase) * D_;
    #pragma unroll 4
    for (int q = 0; q < 64; ++q) {
        const int row = q * 2 + (tid >> 7);
        f32x4 v = *(const f32x4*)(headB + (size_t)row * D_ + kc);
        #pragma unroll
        for (int e = 0; e < 4; ++e) {
            __bf16 h, l2;
            split_bf16(v[e] * u4[e], h, l2);
            sAhi[row][kc + e] = h;
            sAlo[row][kc + e] = l2;
        }
    }
    __syncthreads();   // LDS A read-only from here on

    // Phase 2: sweep O in 4 passes of 128; 8 waves = 4(i) x 2(o), wave 32x64.
    const int lane = tid & 31;
    const int ll   = lane & 15;
    const int g    = lane >> 4;
    const int wave = tid >> 5;
    const int wI   = wave >> 1;            // 0..3
    const int wO   = wave & 1;             // 0..1

    float* __restrict__ outBL = out + ((size_t)b * L_ + l) * (size_t)S_ * S_;

    for (int oPass = 0; oPass < S_ / 128; ++oPass) {
        const int oBase = oPass * 128 + wO * 64;
        v8f acc[2][4] = {};

        #pragma unroll 2
        for (int kt = 0; kt < D_ / TK; ++kt) {
            const int k0 = kt * TK;

            // A fragments (ISA 7.12.2: lane m, K runs 8g.. and 16+8g..).
            v16bf aH[2], aL[2];
            #pragma unroll
            for (int mT = 0; mT < 2; ++mT) {
                const int r = wI * 32 + mT * 16 + ll;
                aH[mT] = CAT16(*(const bf16x8*)&sAhi[r][k0 + 8 * g],
                               *(const bf16x8*)&sAhi[r][k0 + 16 + 8 * g]);
                aL[mT] = CAT16(*(const bf16x8*)&sAlo[r][k0 + 8 * g],
                               *(const bf16x8*)&sAlo[r][k0 + 16 + 8 * g]);
            }

            // B fragments from tiled chunks: wave reads one contiguous 1KB run.
            #pragma unroll
            for (int nT = 0; nT < 4; ++nT) {
                const int oBlk = (oBase >> 4) + nT;
                const __bf16* p = depT
                    + ((size_t)(b * 32 + oBlk) * 16 + kt) * CHUNK
                    + ll * 32 + 8 * g;
                v16bf bHf = CAT16(*(const bf16x8*)p,         *(const bf16x8*)(p + 16));
                v16bf bLf = CAT16(*(const bf16x8*)(p + 512), *(const bf16x8*)(p + 528));
                #pragma unroll
                for (int mT = 0; mT < 2; ++mT) {
                    acc[mT][nT] = WMMA_BF16(aH[mT], bHf, acc[mT][nT]);
                    acc[mT][nT] = WMMA_BF16(aH[mT], bLf, acc[mT][nT]);
                    acc[mT][nT] = WMMA_BF16(aL[mT], bHf, acc[mT][nT]);
                }
            }
        }

        // Store. C/D layout: VGPR r -> M = 8g + r, N = ll.
        #pragma unroll
        for (int mT = 0; mT < 2; ++mT) {
            #pragma unroll
            for (int nT = 0; nT < 4; ++nT) {
                const int n = oBase + nT * 16 + ll;
                #pragma unroll
                for (int r = 0; r < 8; ++r) {
                    const int m = iBase + wI * 32 + mT * 16 + g * 8 + r;
                    outBL[(size_t)m * S_ + n] = acc[mT][nT][r];
                }
            }
        }
    }
}

// ---- fallback (round-1 fused) ------------------------------------------------
constexpr int FTI = 128, FTO = 64, FLDB = 72;

__global__ __launch_bounds__(THREADS)
void bla_fused_kernel(const float* __restrict__ head,
                      const float* __restrict__ dep,
                      const float* __restrict__ U,
                      float* __restrict__ out)
{
    __shared__ __align__(16) __bf16 sAhi[FTI][FLDB];
    __shared__ __align__(16) __bf16 sAlo[FTI][FLDB];
    __shared__ __align__(16) __bf16 sBhi[FTO][FLDB];
    __shared__ __align__(16) __bf16 sBlo[FTO][FLDB];
    __shared__ float sU[D_];

    const int tid = threadIdx.x;
    const int iBase = (blockIdx.x >> 3) * FTI;
    const int oBase = (blockIdx.x & 7) * FTO;
    const int l = blockIdx.y, b = blockIdx.z;

    const float* __restrict__ headB = head + (size_t)b * S_ * D_;
    const float* __restrict__ depB  = dep  + (size_t)b * S_ * D_;

    sU[tid]       = U[l * D_ + tid];
    sU[tid + 256] = U[l * D_ + 256 + tid];

    const int lane = tid & 31, ll = lane & 15, g = lane >> 4;
    const int wave = tid >> 5, wI = wave >> 1, wO = wave & 1;

    v8f acc[2][2] = {};
    for (int kt = 0; kt < D_ / TK; ++kt) {
        const int k0 = kt * TK;
        __syncthreads();
        #pragma unroll
        for (int q = 0; q < 4; ++q) {
            const int idx = q * THREADS + tid;
            const int row = idx >> 3, kc2 = (idx & 7) << 2;
            f32x4 v = *(const f32x4*)(headB + (size_t)(iBase + row) * D_ + k0 + kc2);
            #pragma unroll
            for (int e = 0; e < 4; ++e) {
                __bf16 h, l2;
                split_bf16(v[e] * sU[k0 + kc2 + e], h, l2);
                sAhi[row][kc2 + e] = h; sAlo[row][kc2 + e] = l2;
            }
        }
        #pragma unroll
        for (int q = 0; q < 2; ++q) {
            const int idx = q * THREADS + tid;
            const int row = idx >> 3, kc2 = (idx & 7) << 2;
            f32x4 v = *(const f32x4*)(depB + (size_t)(oBase + row) * D_ + k0 + kc2);
            #pragma unroll
            for (int e = 0; e < 4; ++e) {
                __bf16 h, l2;
                split_bf16(v[e], h, l2);
                sBhi[row][kc2 + e] = h; sBlo[row][kc2 + e] = l2;
            }
        }
        __syncthreads();

        v16bf aH[2], aL[2], bH[2], bL[2];
        #pragma unroll
        for (int mT = 0; mT < 2; ++mT) {
            const int r = wI * 32 + mT * 16 + ll;
            aH[mT] = CAT16(*(const bf16x8*)&sAhi[r][8 * g], *(const bf16x8*)&sAhi[r][16 + 8 * g]);
            aL[mT] = CAT16(*(const bf16x8*)&sAlo[r][8 * g], *(const bf16x8*)&sAlo[r][16 + 8 * g]);
        }
        #pragma unroll
        for (int nT = 0; nT < 2; ++nT) {
            const int r = wO * 32 + nT * 16 + ll;
            bH[nT] = CAT16(*(const bf16x8*)&sBhi[r][8 * g], *(const bf16x8*)&sBhi[r][16 + 8 * g]);
            bL[nT] = CAT16(*(const bf16x8*)&sBlo[r][8 * g], *(const bf16x8*)&sBlo[r][16 + 8 * g]);
        }
        #pragma unroll
        for (int mT = 0; mT < 2; ++mT)
            #pragma unroll
            for (int nT = 0; nT < 2; ++nT) {
                acc[mT][nT] = WMMA_BF16(aH[mT], bH[nT], acc[mT][nT]);
                acc[mT][nT] = WMMA_BF16(aH[mT], bL[nT], acc[mT][nT]);
                acc[mT][nT] = WMMA_BF16(aL[mT], bH[nT], acc[mT][nT]);
            }
    }

    float* __restrict__ outBL = out + ((size_t)b * L_ + l) * (size_t)S_ * S_;
    #pragma unroll
    for (int mT = 0; mT < 2; ++mT)
        #pragma unroll
        for (int nT = 0; nT < 2; ++nT) {
            const int n = oBase + wO * 32 + nT * 16 + ll;
            #pragma unroll
            for (int r = 0; r < 8; ++r) {
                const int m = iBase + wI * 32 + mT * 16 + g * 8 + r;
                outBL[(size_t)m * S_ + n] = acc[mT][nT][r];
            }
        }
}

// ---- launch ------------------------------------------------------------------
extern "C" void kernel_launch(void* const* d_in, const int* in_sizes, int n_in,
                              void* d_out, int out_size, void* d_ws, size_t ws_size,
                              hipStream_t stream) {
    const float* head = (const float*)d_in[0];   // [8, 512, 512]
    const float* dep  = (const float*)d_in[1];   // [8, 512, 512]
    const float* U    = (const float*)d_in[2];   // [32, 512]
    float* out        = (float*)d_out;           // [8, 32, 512, 512]
    (void)in_sizes; (void)n_in; (void)out_size;

    const size_t wsNeed = DEP_ELEMS * 2 * sizeof(__bf16);   // 8 MB tiled hi+lo
    if (ws_size >= wsNeed) {
        __bf16* depT = (__bf16*)d_ws;
        (void)hipFuncSetAttribute((const void*)bla_main_kernel,
                                  hipFuncAttributeMaxDynamicSharedMemorySize,
                                  (int)SMEM_BYTES);
        dep_split_tiled_kernel<<<dim3(B_ * S_ / 16), 256, 0, stream>>>(dep, depT);
        bla_main_kernel<<<dim3(S_ / TI, L_, B_), THREADS, SMEM_BYTES, stream>>>(
            head, U, depT, out);
    } else {
        bla_fused_kernel<<<dim3((S_ / FTI) * (S_ / FTO), L_, B_), THREADS, 0, stream>>>(
            head, dep, U, out);
    }
}
